// PatchedQwen3_5MoeExperts_32384053412430
// MI455X (gfx1250) — matrix operand order repaired and checked by reading, not compiled, not run on hardware
//
#include <hip/hip_runtime.h>
#include <hip/hip_bf16.h>

// ---------------------------------------------------------------------------
// MoE experts (Qwen3.5-style), MI455X / gfx1250, wave32 + WMMA bf16.
// T=2048, H=2048, E=16, I=768, K=2.
// Per 32-token tile of one expert:
//   gu = x @ Wgu^T   (v_wmma_f32_16x16x32_bf16, K over H, B reused by 2 M-tiles)
//   act = silu(gate)*up  (v_exp/v_rcp silu, staged to LDS bf16)
//   y  = act @ Wdown^T   (WMMA, K over I)
//   out[t] += w * y      (global_atomic_add_f32; each token hits 2 experts)
// fp32 weights stream from HBM once (~335MB -> ~14us @ 23.3 TB/s); fp32->bf16
// uses one v_perm_b32 per 2 elements (byte-select the high halves).
// ---------------------------------------------------------------------------

#define TDIM 2048
#define HDIM 2048
#define IDIM 768
#define EDIM 16
#define KSEL 2
#define MAXASSIGN (TDIM * KSEL)    // worst case: everything on one expert
#define MTILE 32
#define NT32 (MAXASSIGN / MTILE)   // 128 token-tiles per expert worst case
#define NWAVES 16                  // 512-thread block, wave32

typedef __attribute__((ext_vector_type(16))) __bf16          v16bf;
typedef __attribute__((ext_vector_type(8)))  float           v8f;
typedef __attribute__((ext_vector_type(8)))  unsigned short  v8us;
typedef __attribute__((ext_vector_type(16))) unsigned short  v16us;
typedef __attribute__((ext_vector_type(8)))  unsigned int    v8ui;

#define LDH 2056   // 2048 + 8 halfs pad (row stride 4112B -> 4-bank skew)
#define LDA 776    // 768 + 8 halfs pad  (row stride 1552B -> 4-bank skew)

// fp32 -> bf16 by truncation (1 shift); accuracy loss <= 1 ulp of bf16.
__device__ __forceinline__ unsigned short f2bf(float f) {
  return (unsigned short)(__builtin_bit_cast(unsigned, f) >> 16);
}

// Pack two fp32 -> bf16x2 in ONE v_perm_b32: dst = {hi16(y), hi16(x)}.
// v_perm_b32 byte indices: 0-3 select S1 bytes, 4-7 select S0 bytes.
__device__ __forceinline__ unsigned pk2(float x, float y) {
  unsigned ux = __builtin_bit_cast(unsigned, x);
  unsigned uy = __builtin_bit_cast(unsigned, y);
#if __has_builtin(__builtin_amdgcn_perm)
  return __builtin_amdgcn_perm(uy, ux, 0x07060302u);  // {y3,y2,x3,x2}
#else
  return (ux >> 16) | (uy & 0xFFFF0000u);
#endif
}

// A fragment (16x32 bf16, MxK) from LDS halfs. k order used here (matches B): lane m (=lane&15)
// holds K = c0..c0+15; c0 = 0 (lanes<16) / 16 (lanes>=16).
__device__ __forceinline__ v16bf load_a_lds(const unsigned short* lds,
                                            int kb, int lane, int stride) {
  const int m  = lane & 15;
  const int c0 = (lane < 16) ? 0 : 16;
  v8us a0 = *(const v8us*)(lds + m * stride + kb + c0);
  v8us a1 = *(const v8us*)(lds + m * stride + kb + c0 + 8);
  v16us r;
#pragma unroll
  for (int i = 0; i < 8; ++i) { r[i] = a0[i]; r[i + 8] = a1[i]; }
  return __builtin_bit_cast(v16bf, r);
}

// B fragment (32x16 bf16, KxN) from 16 contiguous fp32 weights per lane:
// lane n (=lane&15) = column; lanes 0-15 K=kb..kb+15, lanes 16-31 +16
// (caller folds the +0/+16 into p). 8 v_perm_b32 total.
__device__ __forceinline__ v16bf make_b_f32(const float4* p) {
  float4 q0 = p[0], q1 = p[1], q2 = p[2], q3 = p[3];
  v8ui r;
  r[0] = pk2(q0.x, q0.y); r[1] = pk2(q0.z, q0.w);
  r[2] = pk2(q1.x, q1.y); r[3] = pk2(q1.z, q1.w);
  r[4] = pk2(q2.x, q2.y); r[5] = pk2(q2.z, q2.w);
  r[6] = pk2(q3.x, q3.y); r[7] = pk2(q3.z, q3.w);
  return __builtin_bit_cast(v16bf, r);
}

__device__ __forceinline__ v8f wmma_bf16(v16bf a, v16bf b, v8f c) {
  return __builtin_amdgcn_wmma_f32_16x16x32_bf16(false, a, false, b, (short)0,
                                                 c, false, false);
}

// ---------------------------------------------------------------------------
__global__ __launch_bounds__(256) void zero_kernel(float* __restrict__ out,
                                                   int* __restrict__ counts,
                                                   size_t n) {
  size_t i = (size_t)blockIdx.x * blockDim.x + threadIdx.x;
  if (i < n) out[i] = 0.0f;
  if (i < EDIM) counts[i] = 0;
}

__global__ __launch_bounds__(256) void route_kernel(
    const long long* __restrict__ topk_idx, const float* __restrict__ topk_w,
    int* __restrict__ counts, int* __restrict__ tok_list,
    float* __restrict__ w_list) {
  int i = blockIdx.x * blockDim.x + threadIdx.x;   // 0 .. T*K-1
  if (i >= TDIM * KSEL) return;
  int t = i >> 1;                                   // K == 2
  int e = (int)topk_idx[i];
  float w = topk_w[i];
  int pos = atomicAdd(&counts[e], 1);
  tok_list[e * MAXASSIGN + pos] = t;
  w_list[e * MAXASSIGN + pos]  = w;
}

// ---------------------------------------------------------------------------
__global__ __launch_bounds__(512) void moe_expert_kernel(
    const float* __restrict__ hidden,      // [T, H]
    const float* __restrict__ gup,         // [E, 2I, H]
    const float* __restrict__ down,        // [E, H, I]
    const int*   __restrict__ counts,
    const int*   __restrict__ tok_list,
    const float* __restrict__ w_list,
    float*       __restrict__ out) {       // [T, H]
  __shared__ unsigned short lds_h[MTILE * LDH];    // 32-token hidden, bf16
  __shared__ unsigned short lds_act[MTILE * LDA];  // 32-token act, bf16
  __shared__ int   s_tok[MTILE];
  __shared__ float s_w[MTILE];

  const int e    = blockIdx.x;
  const int tile = blockIdx.y;
  const int cnt  = counts[e];
  if (tile * MTILE >= cnt) return;

  const int tid  = threadIdx.x;
  const int lane = tid & 31;
  const int wave = tid >> 5;

  if (tid < MTILE) {
    int idx = tile * MTILE + tid;
    if (idx < cnt) {
      s_tok[tid] = tok_list[e * MAXASSIGN + idx];
      s_w[tid]   = w_list[e * MAXASSIGN + idx];
    } else {
      s_tok[tid] = 0;        // pad row: valid address, weight 0 kills result
      s_w[tid]   = 0.0f;
    }
  }
  __syncthreads();

  // ---- Phase A: stage 32 x H hidden rows into LDS as bf16 ---------------
  // 32*2048 floats = 16384 float4; 512 threads * 32 iters.
#pragma unroll 4
  for (int j = 0; j < 32; ++j) {
    int idx = tid + 512 * j;
    int row = idx >> 9;            // 512 float4 per row
    int c4  = idx & 511;
    float4 v = *(const float4*)(hidden + (size_t)s_tok[row] * HDIM + c4 * 4);
    uint2 p;
    p.x = pk2(v.x, v.y);
    p.y = pk2(v.z, v.w);
    *(uint2*)&lds_h[row * LDH + c4 * 4] = p;
  }
  __syncthreads();

  const int n15  = lane & 15;
  const int kofs = (lane < 16) ? 0 : 16;   // B-fragment K half per lane
  const int mofs = (lane < 16) ? 0 : 8;    // C/D fragment row offset per lane

  // ---- Phase B: gu = x @ Wgu^T, fused silu(gate)*up -> lds_act ----------
  // 48 col-tiles; each of 16 waves owns 3, serving both 16-row M-tiles so
  // each converted B fragment feeds 2 WMMAs.
  for (int g = wave; g < IDIM / 16; g += NWAVES) {
    const int ob = g * 16;
    const float* bg_row =
        gup + ((size_t)e * 2 * IDIM + ob + n15) * HDIM + kofs;
    const float* bu_row =
        gup + ((size_t)e * 2 * IDIM + IDIM + ob + n15) * HDIM + kofs;
    v8f accg0 = {}, accu0 = {}, accg1 = {}, accu1 = {};
    for (int hb = 0; hb < HDIM; hb += 32) {
      __builtin_prefetch(bg_row + hb + 256, 0, 1);   // global_prefetch_b8
      __builtin_prefetch(bu_row + hb + 256, 0, 1);
      v16bf bg = make_b_f32((const float4*)(bg_row + hb));
      v16bf bu = make_b_f32((const float4*)(bu_row + hb));
      v16bf a0 = load_a_lds(lds_h, hb, lane, LDH);
      v16bf a1 = load_a_lds(lds_h + 16 * LDH, hb, lane, LDH);
      accg0 = wmma_bf16(a0, bg, accg0);
      accu0 = wmma_bf16(a0, bu, accu0);
      accg1 = wmma_bf16(a1, bg, accg1);
      accu1 = wmma_bf16(a1, bu, accu1);
    }
#pragma unroll
    for (int v = 0; v < 8; ++v) {
      float g0 = accg0[v], u0 = accu0[v];
      float g1 = accg1[v], u1 = accu1[v];
      float s0 = g0 * __builtin_amdgcn_rcpf(1.0f + __expf(-g0));  // silu
      float s1 = g1 * __builtin_amdgcn_rcpf(1.0f + __expf(-g1));
      lds_act[(v + mofs) * LDA + ob + n15]        = f2bf(s0 * u0);
      lds_act[(v + mofs + 16) * LDA + ob + n15]   = f2bf(s1 * u1);
    }
  }
  __syncthreads();

  // ---- Phase C: y = act @ Wdown^T, scaled atomic accumulate -------------
  float wv0[8], wv1[8];
  int   tk0[8], tk1[8];
#pragma unroll
  for (int v = 0; v < 8; ++v) {
    wv0[v] = s_w[v + mofs];        tk0[v] = s_tok[v + mofs];
    wv1[v] = s_w[v + mofs + 16];   tk1[v] = s_tok[v + mofs + 16];
  }

  // 128 output h-tiles of 16; each wave owns 8, both M-tiles per B fragment.
  for (int nt = wave; nt < HDIM / 16; nt += NWAVES) {
    const int hb16 = nt * 16;
    const float* b_row =
        down + ((size_t)e * HDIM + hb16 + n15) * IDIM + kofs;
    v8f acc0 = {}, acc1 = {};
    for (int ib = 0; ib < IDIM; ib += 32) {
      __builtin_prefetch(b_row + ib + 128, 0, 1);
      v16bf b  = make_b_f32((const float4*)(b_row + ib));
      v16bf a0 = load_a_lds(lds_act, ib, lane, LDA);
      v16bf a1 = load_a_lds(lds_act + 16 * LDA, ib, lane, LDA);
      acc0 = wmma_bf16(a0, b, acc0);
      acc1 = wmma_bf16(a1, b, acc1);
    }
    const int col = hb16 + n15;
#pragma unroll
    for (int v = 0; v < 8; ++v) {
      atomicAdd(out + (size_t)tk0[v] * HDIM + col, wv0[v] * acc0[v]);
      atomicAdd(out + (size_t)tk1[v] * HDIM + col, wv1[v] * acc1[v]);
    }
  }
}

// ---------------------------------------------------------------------------
extern "C" void kernel_launch(void* const* d_in, const int* in_sizes, int n_in,
                              void* d_out, int out_size, void* d_ws,
                              size_t ws_size, hipStream_t stream) {
  const float*     hidden   = (const float*)d_in[0];
  const long long* topk_idx = (const long long*)d_in[1];  // int64 per reference
  const float*     topk_w   = (const float*)d_in[2];
  const float*     gup      = (const float*)d_in[3];
  const float*     down     = (const float*)d_in[4];
  float*           out      = (float*)d_out;

  // Workspace layout: counts[16] | tok_list[16][4096] | w_list[16][4096]
  int*   counts   = (int*)d_ws;
  int*   tok_list = counts + EDIM;
  float* w_list   = (float*)(tok_list + EDIM * MAXASSIGN);

  const size_t n_out = (size_t)TDIM * HDIM;
  zero_kernel<<<(unsigned)((n_out + 255) / 256), 256, 0, stream>>>(out, counts,
                                                                   n_out);
  route_kernel<<<(TDIM * KSEL + 255) / 256, 256, 0, stream>>>(
      topk_idx, topk_w, counts, tok_list, w_list);

  dim3 grid(EDIM, NT32);     // blocks past counts[e] early-exit
  moe_expert_kernel<<<grid, 512, 0, stream>>>(hidden, gup, down, counts,
                                              tok_list, w_list, out);
}
